// DeepFMRanker_56710748176670
// MI455X (gfx1250) — matrix operand order repaired and checked by reading, not compiled, not run on hardware
//
#include <hip/hip_runtime.h>

typedef __attribute__((ext_vector_type(16))) _Float16 v16h;
typedef __attribute__((ext_vector_type(8)))  _Float16 v8h;
typedef __attribute__((ext_vector_type(8)))  float    v8f;

#define B_ROWS  16384
#define D_EMB   32
#define NGENRE  6
#define ND      8
#define K_RAW   200
#define DEEP_K  224      // 200 padded to 7 * 32 for f16 WMMA K-tiling
#define H1      128
#define H2      64

// ---------------------------------------------------------------------------
// Kernel 1: fused gather + FM first/second order + deep_in (f16) assembly.
// One wave32 per batch row; lane == embedding dim (D_EMB == 32 == warpSize).
// ---------------------------------------------------------------------------
__global__ void prep_kernel(
    const int* __restrict__ user_id,   const int* __restrict__ item_id,
    const int* __restrict__ gender,    const int* __restrict__ age,
    const int* __restrict__ occ,       const int* __restrict__ genre_ids,
    const float* __restrict__ genre_mask, const float* __restrict__ dense,
    const float* __restrict__ fo_user, const float* __restrict__ fo_item,
    const float* __restrict__ fo_gender, const float* __restrict__ fo_age,
    const float* __restrict__ fo_occ,  const float* __restrict__ fo_genre,
    const float* __restrict__ emb_user, const float* __restrict__ emb_item,
    const float* __restrict__ emb_gender, const float* __restrict__ emb_age,
    const float* __restrict__ emb_occ, const float* __restrict__ emb_genre,
    const float* __restrict__ dense_W, const float* __restrict__ dense_b,
    _Float16* __restrict__ deep_in, float* __restrict__ fs)
{
    const int wave = (blockIdx.x * blockDim.x + threadIdx.x) >> 5;
    const int lane = threadIdx.x & 31;
    if (wave >= B_ROWS) return;
    const int b = wave;

    const int u  = user_id[b];
    const int it = item_id[b];
    const int g  = gender[b];
    const int a  = age[b];
    const int oc = occ[b];

    // per-lane embedding dim gathers
    float xu = emb_user  [(size_t)u  * D_EMB + lane];
    float xi = emb_item  [(size_t)it * D_EMB + lane];
    float xg = emb_gender[(size_t)g  * D_EMB + lane];
    float xa = emb_age   [(size_t)a  * D_EMB + lane];
    float xo = emb_occ   [(size_t)oc * D_EMB + lane];

    float msum = 0.f, ep = 0.f, fop = 0.f;
    #pragma unroll
    for (int j = 0; j < NGENRE; ++j) {
        float m  = genre_mask[b * NGENRE + j];
        int  gid = genre_ids [b * NGENRE + j];
        msum += m;
        ep   += m * emb_genre[(size_t)gid * D_EMB + lane];
        fop  += m * fo_genre[gid];
    }
    const float denom = fmaxf(msum, 1.0f);
    ep /= denom;

    // FM second order, per-dim contribution then cross-lane sum
    const float s  = xu + xi + xg + xa + xo + ep;
    const float sq = xu*xu + xi*xi + xg*xg + xa*xa + xo*xo + ep*ep;
    float part = 0.5f * (s * s - sq);
    #pragma unroll
    for (int off = 16; off >= 1; off >>= 1)
        part += __shfl_xor(part, off, 32);

    // deep_in row: [6*32 field embeds | 8 dense | 24 zero pad] as f16
    const size_t base = (size_t)b * DEEP_K;
    deep_in[base + 0*D_EMB + lane] = (_Float16)xu;
    deep_in[base + 1*D_EMB + lane] = (_Float16)xi;
    deep_in[base + 2*D_EMB + lane] = (_Float16)xg;
    deep_in[base + 3*D_EMB + lane] = (_Float16)xa;
    deep_in[base + 4*D_EMB + lane] = (_Float16)xo;
    deep_in[base + 5*D_EMB + lane] = (_Float16)ep;
    // lanes 0..7 -> dense[192..199], lanes 8..31 -> zero pad [200..223]
    deep_in[base + 192 + lane] =
        (lane < ND) ? (_Float16)dense[b * ND + lane] : (_Float16)0.f;

    if (lane == 0) {
        float first = fo_user[u] + fo_item[it] + fo_gender[g] + fo_age[a]
                    + fo_occ[oc] + fop / denom + dense_b[0];
        #pragma unroll
        for (int k = 0; k < ND; ++k) first += dense[b * ND + k] * dense_W[k];
        fs[b] = first + part;   // part already holds full 32-lane sum
    }
}

// ---------------------------------------------------------------------------
// Weight conversion f32 -> f16 (W1 padded K 200 -> 224 with zeros).
// Tiny; re-run each launch for determinism.
// ---------------------------------------------------------------------------
__global__ void convW1_kernel(const float* __restrict__ W1, _Float16* __restrict__ W1h) {
    int idx = blockIdx.x * blockDim.x + threadIdx.x;
    if (idx >= H1 * DEEP_K) return;
    int n = idx / DEEP_K, k = idx % DEEP_K;
    W1h[idx] = (k < K_RAW) ? (_Float16)W1[n * K_RAW + k] : (_Float16)0.f;
}
__global__ void convW2_kernel(const float* __restrict__ W2, _Float16* __restrict__ W2h) {
    int idx = blockIdx.x * blockDim.x + threadIdx.x;
    if (idx >= H2 * H1) return;
    W2h[idx] = (_Float16)W2[idx];
}

// ---------------------------------------------------------------------------
// WMMA GEMM: C[M,N] = relu(A[M,K] * Bw[N,K]^T + bias), f16 in/out, f32 acc.
// One wave per 16x16 output tile; K stepped in 32s via v_wmma_f32_16x16x32_f16.
// Fragment layouts per CDNA5 ISA 7.12.2 (wave32).
// ---------------------------------------------------------------------------
template<int N, int K, bool RELU>
__global__ void gemm_wmma_kernel(const _Float16* __restrict__ A,
                                 const _Float16* __restrict__ Bw,   // [N, K]
                                 const float*    __restrict__ bias, // [N]
                                 _Float16*       __restrict__ C,    // [M, N]
                                 int M)
{
    const int lane       = threadIdx.x & 31;
    const int waveInBlk  = threadIdx.x >> 5;
    const int wavesPerBlk = blockDim.x >> 5;
    const int tile = blockIdx.x * wavesPerBlk + waveInBlk;
    const int ntn  = N / 16;
    const int tm   = tile / ntn;
    const int tn   = tile % ntn;
    if (tm * 16 >= M) return;                 // wave-uniform: EXEC stays all-1s

    const int halfSel = lane >> 4;            // 0: lanes 0-15, 1: lanes 16-31
    const int l15     = lane & 15;

    const _Float16* arow = A  + (size_t)(tm * 16 + l15) * K;  // A row (M = l15)
    const _Float16* brow = Bw + (size_t)(tn * 16 + l15) * K;  // weight row (N = l15)

    v8f acc = {};
    #pragma unroll
    for (int kt = 0; kt < K; kt += 32) {
        // A 16x32 f16 fragment: elems 0-7 = K kt+8*halfSel.., elems 8-15 = +16
        v8h a0 = *(const v8h*)(arow + kt + halfSel * 8);
        v8h a1 = *(const v8h*)(arow + kt + 16 + halfSel * 8);
        // B 32x16 f16 fragment: elems 0-7 = K kt+8*halfSel.., elems 8-15 = +16
        v8h b0 = *(const v8h*)(brow + kt + halfSel * 8);
        v8h b1 = *(const v8h*)(brow + kt + 16 + halfSel * 8);
        v16h af, bf;
        #pragma unroll
        for (int e = 0; e < 8; ++e) {
            af[e] = a0[e]; af[e + 8] = a1[e];
            bf[e] = b0[e]; bf[e + 8] = b1[e];
        }
        acc = __builtin_amdgcn_wmma_f32_16x16x32_f16(
                  false, af, false, bf, (short)0, acc, false, false);
    }

    // Epilogue: D VGPR v -> M = v + 8*halfSel, N = l15 (fixed per lane)
    const int n  = tn * 16 + l15;
    const float bv = bias[n];
    #pragma unroll
    for (int v = 0; v < 8; ++v) {
        const int m = tm * 16 + v + halfSel * 8;
        float r = acc[v] + bv;
        if (RELU) r = fmaxf(r, 0.f);
        C[(size_t)m * N + n] = (_Float16)r;
    }
}

// ---------------------------------------------------------------------------
// Final: out[b] = fs[b] + h2[b,:] . Wout + bout
// ---------------------------------------------------------------------------
__global__ void final_kernel(const _Float16* __restrict__ h2,
                             const float* __restrict__ Wout,
                             const float* __restrict__ bout,
                             const float* __restrict__ fs,
                             float* __restrict__ out)
{
    int b = blockIdx.x * blockDim.x + threadIdx.x;
    if (b >= B_ROWS) return;
    float acc = bout[0];
    const _Float16* row = h2 + (size_t)b * H2;
    #pragma unroll
    for (int k = 0; k < H2; ++k) acc += (float)row[k] * Wout[k];
    out[b] = fs[b] + acc;
}

// ---------------------------------------------------------------------------
extern "C" void kernel_launch(void* const* d_in, const int* in_sizes, int n_in,
                              void* d_out, int out_size, void* d_ws, size_t ws_size,
                              hipStream_t stream) {
    (void)in_sizes; (void)n_in; (void)out_size; (void)ws_size;

    const int*   user_id    = (const int*)  d_in[0];
    const int*   item_id    = (const int*)  d_in[1];
    const int*   gender     = (const int*)  d_in[2];
    const int*   age        = (const int*)  d_in[3];
    const int*   occ        = (const int*)  d_in[4];
    const int*   genre_ids  = (const int*)  d_in[5];
    const float* genre_mask = (const float*)d_in[6];
    const float* dense      = (const float*)d_in[7];
    const float* fo_user    = (const float*)d_in[8];
    const float* fo_item    = (const float*)d_in[9];
    const float* fo_gender  = (const float*)d_in[10];
    const float* fo_age     = (const float*)d_in[11];
    const float* fo_occ     = (const float*)d_in[12];
    const float* fo_genre   = (const float*)d_in[13];
    const float* emb_user   = (const float*)d_in[14];
    const float* emb_item   = (const float*)d_in[15];
    const float* emb_gender = (const float*)d_in[16];
    const float* emb_age    = (const float*)d_in[17];
    const float* emb_occ    = (const float*)d_in[18];
    const float* emb_genre  = (const float*)d_in[19];
    const float* dense_W    = (const float*)d_in[20];
    const float* dense_b    = (const float*)d_in[21];
    const float* W1         = (const float*)d_in[22];
    const float* b1         = (const float*)d_in[23];
    const float* W2         = (const float*)d_in[24];
    const float* b2         = (const float*)d_in[25];
    const float* Wout       = (const float*)d_in[26];
    const float* bout       = (const float*)d_in[27];
    float* out = (float*)d_out;

    // Workspace layout (all 16B-aligned for B128 fragment loads)
    char* ws = (char*)d_ws;
    _Float16* deep_in = (_Float16*)(ws);                              // 16384*224*2 = 7340032
    _Float16* h1      = (_Float16*)(ws + 7340032);                    // 16384*128*2 = 4194304
    _Float16* h2      = (_Float16*)(ws + 7340032 + 4194304);          // 16384*64*2  = 2097152
    float*    fs      = (float*)   (ws + 7340032 + 4194304 + 2097152);          // 65536
    _Float16* W1h     = (_Float16*)(ws + 7340032 + 4194304 + 2097152 + 65536);  // 57344
    _Float16* W2h     = (_Float16*)(ws + 7340032 + 4194304 + 2097152 + 65536 + 57344); // 16384

    // Weight conversions
    convW1_kernel<<<(H1 * DEEP_K + 255) / 256, 256, 0, stream>>>(W1, W1h);
    convW2_kernel<<<(H2 * H1 + 255) / 256, 256, 0, stream>>>(W2, W2h);

    // Fused gathers + FM terms + deep_in assembly: 1 wave/row, 8 waves/block
    prep_kernel<<<B_ROWS / 8, 256, 0, stream>>>(
        user_id, item_id, gender, age, occ, genre_ids, genre_mask, dense,
        fo_user, fo_item, fo_gender, fo_age, fo_occ, fo_genre,
        emb_user, emb_item, emb_gender, emb_age, emb_occ, emb_genre,
        dense_W, dense_b, deep_in, fs);

    // GEMM1: [16384,224] x [128,224]^T -> relu -> h1 [16384,128]
    {
        const int tiles = (B_ROWS / 16) * (H1 / 16);   // 1024 * 8 = 8192
        gemm_wmma_kernel<H1, DEEP_K, true><<<tiles / 4, 128, 0, stream>>>(
            deep_in, W1h, b1, h1, B_ROWS);
    }
    // GEMM2: [16384,128] x [64,128]^T -> relu -> h2 [16384,64]
    {
        const int tiles = (B_ROWS / 16) * (H2 / 16);   // 1024 * 4 = 4096
        gemm_wmma_kernel<H2, H1, true><<<tiles / 4, 128, 0, stream>>>(
            h1, W2h, b2, h2, B_ROWS);
    }
    // Final dot + combine
    final_kernel<<<B_ROWS / 256, 256, 0, stream>>>(h2, Wout, bout, fs, out);
}
